// KarmaBlock_7335804142288
// MI455X (gfx1250) — matrix operand order repaired and checked by reading, not compiled, not run on hardware
//
#include <hip/hip_runtime.h>

// ---------------------------------------------------------------------------
// KarmaBlock for gfx1250: 4x Mamba (time fwd/bwd @512, two freq @259).
// GEMMs: v_wmma_f32_16x16x32_bf16, A tiles streamed to LDS with
// global_load_async_to_lds_b128 (double-buffered, s_wait_asynccnt),
// B fragments from pre-transposed padded bf16 weights.
// ---------------------------------------------------------------------------

typedef __bf16 bf16_t;
typedef bf16_t v16bf __attribute__((ext_vector_type(16)));
typedef float  v8f   __attribute__((ext_vector_type(8)));

#define MTOT 4096   // B*L
#define LSEQ 1024
#define BB   4
#define TILE_K 32
#define TILE_BYTES (16 * TILE_K * 2)   // 1024 B per A tile

static inline int pad16(int x) { return (x + 15) & ~15; }
static inline int pad32(int x) { return (x + 31) & ~31; }

// ---------------------------------------------------------------------------
// Async copy of one 16x32 bf16 A tile (1 KB) global->LDS: two
// global_load_async_to_lds_b128 ops, 32 lanes x 16 B each (rows 0-7, 8-15).
// Issued by wave 0 only; tracked by ASYNCcnt.
// ---------------------------------------------------------------------------
__device__ __forceinline__ void async_tile_16x32(uint32_t lds_lane,
                                                 const bf16_t* g_lane, int Kp) {
  asm volatile("global_load_async_to_lds_b128 %0, %1, off"
               :: "v"(lds_lane), "v"(g_lane) : "memory");
  asm volatile("global_load_async_to_lds_b128 %0, %1, off"
               :: "v"(lds_lane + 512u), "v"(g_lane + (size_t)8 * Kp) : "memory");
}

// ---------------------------------------------------------------------------
// Weight pre-pass: W[K,N] f32 -> Wt[Np,Kp] bf16 transposed, zero padded.
// Pair-packed u32 stores.
// ---------------------------------------------------------------------------
__global__ void k_convert_wt(const float* __restrict__ W, bf16_t* __restrict__ Wt,
                             int K, int N, int Kp, int Np) {
  long idx = (long)blockIdx.x * blockDim.x + threadIdx.x;
  long total2 = (long)Np * (Kp >> 1);
  if (idx >= total2) return;
  int kh = (int)(idx % (Kp >> 1));
  int n  = (int)(idx / (Kp >> 1));
  int k  = kh * 2;
  float a0 = (n < N && k     < K) ? W[(size_t)k * N + n]       : 0.0f;
  float a1 = (n < N && k + 1 < K) ? W[(size_t)(k + 1) * N + n] : 0.0f;
  union { bf16_t h[2]; unsigned u; } pk;
  pk.h[0] = (bf16_t)a0; pk.h[1] = (bf16_t)a1;
  ((unsigned*)Wt)[idx] = pk.u;
}

// ---------------------------------------------------------------------------
// Activation pre-pass: A[M,K] f32 (row stride lda) -> ABf[M,Kp] bf16 padded.
// ---------------------------------------------------------------------------
__global__ void k_convert_act(const float* __restrict__ A, int lda, int K, int Kp,
                              bf16_t* __restrict__ ABf, long total2) {
  long idx = (long)blockIdx.x * blockDim.x + threadIdx.x;
  if (idx >= total2) return;
  int  kh = (int)(idx % (Kp >> 1));
  long r  = idx / (Kp >> 1);
  int  k  = kh * 2;
  float a0 = (k     < K) ? A[(size_t)r * lda + k]     : 0.0f;
  float a1 = (k + 1 < K) ? A[(size_t)r * lda + k + 1] : 0.0f;
  union { bf16_t h[2]; unsigned u; } pk;
  pk.h[0] = (bf16_t)a0; pk.h[1] = (bf16_t)a1;
  ((unsigned*)ABf)[idx] = pk.u;
}

// ---------------------------------------------------------------------------
// WMMA GEMM: C[M,N] = ABf[M,Kp](bf16) * Wt[Np,Kp](bf16 transposed), f32 acc.
// 256 threads = 8 waves; block = 16 rows x 128 cols; wave w owns one 16x16
// tile. A tile double-buffered in LDS via async copy (wave 0 producer).
// ---------------------------------------------------------------------------
__global__ __launch_bounds__(256)
void k_gemm(const bf16_t* __restrict__ ABf, int Kp,
            const bf16_t* __restrict__ Wt,
            float* __restrict__ C, int ldc, int N, int Np) {
  __shared__ __align__(16) bf16_t As[2][16 * TILE_K];

  const int m0   = blockIdx.x * 16;
  const int wave = threadIdx.x >> 5;
  const int lane = threadIdx.x & 31;
  const int mloc = lane & 15;
  const int half = lane >> 4;
  const int n0   = blockIdx.y * 128 + wave * 16;
  const bool active = (n0 < Np);

  // producer lane mapping: row = lane>>2 (0..7), 16B segment = lane&3
  const int arow = lane >> 2;
  const int aseg = lane & 3;
  const uint32_t lds_lane0 =
      (uint32_t)(uintptr_t)(&As[0][0]) + (uint32_t)(arow * 64 + aseg * 16);
  const bf16_t* g_lane0 = ABf + (size_t)(m0 + arow) * Kp + aseg * 8;

  const int T = Kp / TILE_K;
  if (wave == 0) async_tile_16x32(lds_lane0, g_lane0, Kp);   // prologue: tile 0

  const bf16_t* wp = Wt + (size_t)(active ? (n0 + mloc) : 0) * Kp + half * 8;
  v8f acc = {};

  for (int kt = 0; kt < T; ++kt) {
    if (wave == 0) {
      if (kt + 1 < T) {
        async_tile_16x32(lds_lane0 + (uint32_t)(((kt + 1) & 1) * TILE_BYTES),
                         g_lane0 + (size_t)(kt + 1) * TILE_K, Kp);
        asm volatile("s_wait_asynccnt 0x2" ::: "memory");  // tile kt landed
      } else {
        asm volatile("s_wait_asynccnt 0x0" ::: "memory");  // last tile landed
      }
    }
    __syncthreads();
    if (active) {
      union { uint4 q[2]; v16bf v; } ua, ub;
      // A frag (ISA 7.12.2): row mloc, chunks [half*8..+7], [16+half*8..+7]
      const bf16_t* ap = &As[kt & 1][mloc * TILE_K + half * 8];
      ua.q[0] = *(const uint4*)(ap);
      ua.q[1] = *(const uint4*)(ap + 16);
      // B frag: column n0+mloc, chunks [half*8..+7], [16+half*8..+7]
      const bf16_t* bp = wp + kt * TILE_K;
      ub.q[0] = *(const uint4*)(bp);
      ub.q[1] = *(const uint4*)(bp + 16);
      __builtin_prefetch(bp + TILE_K, 0, 1);   // global_prefetch_b8
      acc = __builtin_amdgcn_wmma_f32_16x16x32_bf16(
          false, ua.v, false, ub.v, (short)0, acc, false, false);
    }
    __syncthreads();   // protect buffer (kt&1) until tile kt+2 is issued
  }

  if (active) {
    const int col = n0 + mloc;
    if (col < N) {
#pragma unroll
      for (int r = 0; r < 8; ++r)   // C/D layout: VGPR r -> row r + 8*half
        C[(size_t)(m0 + r + half * 8) * ldc + col] = acc[r];
    }
  }
}

// ---------------------------------------------------------------------------
// RMSNorm per token row, optional time flip on output.
// ---------------------------------------------------------------------------
__global__ __launch_bounds__(256)
void k_rmsnorm(const float* __restrict__ x, const float* __restrict__ w,
               float* __restrict__ out, int D, int L_, int reverse) {
  __shared__ float red[256];
  const int row = blockIdx.x;
  const float* xr = x + (size_t)row * D;
  float s = 0.0f;
  for (int c = threadIdx.x; c < D; c += 256) { float v = xr[c]; s += v * v; }
  red[threadIdx.x] = s;
  __syncthreads();
  for (int off = 128; off > 0; off >>= 1) {
    if (threadIdx.x < off) red[threadIdx.x] += red[threadIdx.x + off];
    __syncthreads();
  }
  float rs = rsqrtf(red[0] / (float)D + 1e-5f);
  int b = row / L_, l = row % L_;
  int orow = b * L_ + (reverse ? (L_ - 1 - l) : l);
  float* o = out + (size_t)orow * D;
  for (int c = threadIdx.x; c < D; c += 256) o[c] = xr[c] * rs * w[c];
}

// ---------------------------------------------------------------------------
// Causal depthwise conv (k=4) + bias + SiLU, reading xi half of xz[M,2*di].
// ---------------------------------------------------------------------------
__global__ void k_conv_silu(const float* __restrict__ xz, const float* __restrict__ cw,
                            const float* __restrict__ cb, float* __restrict__ xi,
                            int di, int L_, long total) {
  long idx = (long)blockIdx.x * blockDim.x + threadIdx.x;
  if (idx >= total) return;
  int d = (int)(idx % di);
  long row = idx / di;
  int l = (int)(row % L_);
  const size_t ld2 = (size_t)2 * di;
  float acc = cb[d];
#pragma unroll
  for (int j = 0; j < 4; ++j) {
    int l2 = l - 3 + j;
    if (l2 >= 0) acc += cw[d * 4 + j] * xz[(size_t)(row - 3 + j) * ld2 + d];
  }
  float sig = 1.0f / (1.0f + __expf(-acc));
  xi[idx] = acc * sig;
}

// ---------------------------------------------------------------------------
// Selective scan: block per batch, thread per channel, 16 states in registers,
// B_t/C_t broadcast via LDS. y written in-place over xi.
// ---------------------------------------------------------------------------
__global__ __launch_bounds__(1024)
void k_scan(float* __restrict__ xi, const float* __restrict__ dtraw,
            const float* __restrict__ xdbl, const float* __restrict__ dt_bias,
            const float* __restrict__ A_log, const float* __restrict__ Dp,
            int L_, int di, int ndbl, int dtr) {
  __shared__ float sBC[32];
  const int b = blockIdx.x;
  const int d = threadIdx.x;
  float A[16], h[16];
  float Dd = 0.0f, bias = 0.0f;
#pragma unroll
  for (int n = 0; n < 16; ++n) { A[n] = 0.0f; h[n] = 0.0f; }
  if (d < di) {
    Dd = Dp[d]; bias = dt_bias[d];
#pragma unroll
    for (int n = 0; n < 16; ++n) A[n] = -__expf(A_log[d * 16 + n]);
  }
  for (int l = 0; l < L_; ++l) {
    size_t row = (size_t)b * L_ + l;
    if (threadIdx.x < 32) sBC[threadIdx.x] = xdbl[row * ndbl + dtr + threadIdx.x];
    __syncthreads();
    if (d < di) {
      float dtv = dtraw[row * di + d] + bias;
      dtv = (dtv > 20.0f) ? dtv : log1pf(__expf(dtv));   // softplus
      float xv = xi[row * di + d];
      float dx = dtv * xv;
      float acc = 0.0f;
#pragma unroll
      for (int n = 0; n < 16; ++n) {
        h[n] = h[n] * __expf(dtv * A[n]) + dx * sBC[n];
        acc += h[n] * sBC[16 + n];
      }
      xi[row * di + d] = acc + Dd * xv;
    }
    __syncthreads();
  }
}

// y *= silu(z), z from second half of xz.
__global__ void k_gate(float* __restrict__ y, const float* __restrict__ xz,
                       int di, long total) {
  long idx = (long)blockIdx.x * blockDim.x + threadIdx.x;
  if (idx >= total) return;
  int d = (int)(idx % di);
  long row = idx / di;
  float z = xz[(size_t)row * (2 * di) + di + d];
  y[idx] *= z / (1.0f + __expf(-z));
}

// x_time = out1 + flip_L(out2) + x
__global__ void k_combine(const float* __restrict__ o1, const float* __restrict__ o2,
                          const float* __restrict__ x, float* __restrict__ out,
                          int L_, int D, long total) {
  long idx = (long)blockIdx.x * blockDim.x + threadIdx.x;
  if (idx >= total) return;
  int c = (int)(idx % D);
  long row = idx / D;
  int b = (int)(row / L_), l = (int)(row % L_);
  long rrow = (long)b * L_ + (L_ - 1 - l);
  out[idx] = o1[idx] + o2[rrow * D + c] + x[idx];
}

// ---------------------------------------------------------------------------
// Host-side branch driver
// ---------------------------------------------------------------------------
struct MambaParams {
  const float *in_proj, *conv_w, *conv_b, *x_proj, *dt_proj, *dt_bias, *A_log, *D, *out_proj;
  int dmodel, di, dtr;
};

static void gemm_stage(const float* act, int lda, int K, const float* W, int N,
                       float* C, int ldc, bf16_t* ABf, bf16_t* WT, hipStream_t s) {
  const int M = MTOT;
  const int Kp = pad32(K), Np = pad16(N);
  long tw = (long)Np * (Kp >> 1);
  k_convert_wt<<<(int)((tw + 255) / 256), 256, 0, s>>>(W, WT, K, N, Kp, Np);
  long ta = (long)M * (Kp >> 1);
  k_convert_act<<<(int)((ta + 255) / 256), 256, 0, s>>>(act, lda, K, Kp, ABf, ta);
  dim3 g(M / 16, (Np + 127) / 128);
  k_gemm<<<g, 256, 0, s>>>(ABf, Kp, WT, C, ldc, N, Np);
}

static void run_mamba(const MambaParams& p, const float* input, float* out, int ldout,
                      float* XZ, float* XI, float* XD, float* DT,
                      bf16_t* ABf, bf16_t* WT, hipStream_t s) {
  const int M = MTOT, Lc = LSEQ;
  const int d2 = 2 * p.di, ndbl = p.dtr + 32;

  // 1) xz = input @ in_proj
  gemm_stage(input, p.dmodel, p.dmodel, p.in_proj, d2, XZ, d2, ABf, WT, s);

  // 2) xi = silu(causal_conv(xz[:, :di]) + b)
  { long tot = (long)M * p.di;
    k_conv_silu<<<(int)((tot + 255) / 256), 256, 0, s>>>(XZ, p.conv_w, p.conv_b, XI, p.di, Lc, tot); }

  // 3) x_dbl = xi @ x_proj
  gemm_stage(XI, p.di, p.di, p.x_proj, ndbl, XD, ndbl, ABf, WT, s);

  // 4) dt_raw = x_dbl[:, :dtr] @ dt_proj   (bias+softplus folded into scan)
  gemm_stage(XD, ndbl, p.dtr, p.dt_proj, p.di, DT, p.di, ABf, WT, s);

  // 5) selective scan (y in-place over XI)
  { int bdim = ((p.di + 31) / 32) * 32;
    k_scan<<<BB, bdim, 0, s>>>(XI, DT, XD, p.dt_bias, p.A_log, p.D, Lc, p.di, ndbl, p.dtr); }

  // 6) y *= silu(z)
  { long tot = (long)M * p.di;
    k_gate<<<(int)((tot + 255) / 256), 256, 0, s>>>(XI, XZ, p.di, tot); }

  // 7) out = y @ out_proj
  gemm_stage(XI, p.di, p.di, p.out_proj, p.dmodel, out, ldout, ABf, WT, s);
}

extern "C" void kernel_launch(void* const* d_in, const int* in_sizes, int n_in,
                              void* d_out, int out_size, void* d_ws, size_t ws_size,
                              hipStream_t stream) {
  (void)in_sizes; (void)n_in; (void)out_size; (void)ws_size;
  const int M = MTOT;

  const float* x   = (const float*)d_in[0];
  const float* x1  = (const float*)d_in[1];
  const float* x2  = (const float*)d_in[2];
  const float* wn  = (const float*)d_in[39];
  const float* wn1 = (const float*)d_in[40];

  auto getP = [&](int base, int dm) {
    MambaParams p;
    p.in_proj  = (const float*)d_in[base + 0];
    p.conv_w   = (const float*)d_in[base + 1];
    p.conv_b   = (const float*)d_in[base + 2];
    p.x_proj   = (const float*)d_in[base + 3];
    p.dt_proj  = (const float*)d_in[base + 4];
    p.dt_bias  = (const float*)d_in[base + 5];
    p.A_log    = (const float*)d_in[base + 6];
    p.D        = (const float*)d_in[base + 7];
    p.out_proj = (const float*)d_in[base + 8];
    p.dmodel = dm; p.di = 2 * dm; p.dtr = (dm + 15) / 16;
    return p;
  };
  MambaParams pt1 = getP(3, 512), pt2 = getP(12, 512);
  MambaParams pf1 = getP(21, 259), pf2 = getP(30, 259);

  float* out_time = (float*)d_out;                 // [M, 512]
  float* out_f1   = out_time + (size_t)M * 512;    // [M, 259]
  float* out_f2   = out_f1 + (size_t)M * 259;      // [M, 259]

  // workspace layout (floats unless noted); peak ~104 MB
  float* ws = (float*)d_ws;
  float* XZ = ws;                          // M*2048
  float* XI = XZ + (size_t)M * 2048;       // M*1024
  float* XD = XI + (size_t)M * 1024;       // M*64
  float* DT = XD + (size_t)M * 64;         // M*1024
  float* XN = DT + (size_t)M * 1024;       // M*512 (normalized input, reused)
  float* O1 = XN + (size_t)M * 512;        // M*512
  float* O2 = O1 + (size_t)M * 512;        // M*512
  bf16_t* WT  = (bf16_t*)(O2 + (size_t)M * 512);     // <= 2048*512 bf16 (2 MB)
  bf16_t* ABf = WT + (size_t)2048 * 512;             // M*1024 bf16 (8 MB)

  // freq branches (direct to d_out)
  run_mamba(pf1, x1, out_f1, 259, XZ, XI, XD, DT, ABf, WT, stream);
  run_mamba(pf2, x2, out_f2, 259, XZ, XI, XD, DT, ABf, WT, stream);

  // time forward
  k_rmsnorm<<<M, 256, 0, stream>>>(x, wn, XN, 512, LSEQ, 0);
  run_mamba(pt1, XN, O1, 512, XZ, XI, XD, DT, ABf, WT, stream);

  // time backward (input pre-flipped, output un-flipped in combine)
  k_rmsnorm<<<M, 256, 0, stream>>>(x, wn1, XN, 512, LSEQ, 1);
  run_mamba(pt2, XN, O2, 512, XZ, XI, XD, DT, ABf, WT, stream);

  long tot = (long)M * 512;
  k_combine<<<(int)((tot + 255) / 256), 256, 0, stream>>>(O1, O2, x, out_time, LSEQ, 512, tot);
}
